// vFusedLiveSR_2559800508453
// MI455X (gfx1250) — matrix operand order repaired and checked by reading, not compiled, not run on hardware
//
#include <hip/hip_runtime.h>

typedef __attribute__((ext_vector_type(16))) _Float16 v16h;
typedef __attribute__((ext_vector_type(8)))  _Float16 v8h;
typedef __attribute__((ext_vector_type(8)))  float    v8f;

// ---------------- problem constants ----------------
constexpr int NIMG = 2048;
constexpr int F    = 64;       // features
constexpr int CAP  = 768;

// ---------------- ws layout (bytes) ----------------
constexpr size_t WS_COUNT    = 0;       // int
constexpr size_t WS_WORKLIST = 16;      // int[2048] packed (img<<3)|e
constexpr size_t WS_EXPERT   = 12288;   // int[2048]
constexpr size_t WS_WHALF    = 20480;   // _Float16 fragment-packed weights

// half-index layout inside weights region (each fragment = 512 halves = 1KB)
constexpr int HF_HEAD        = 0;
constexpr int HF_HEAD_SZ     = 4 * 4 * 512;          // 4 experts x 4 N-frags
constexpr int HF_BODY        = HF_HEAD_SZ;           // 8192
constexpr int HF_BODY_LAYER  = 72 * 512;             // 9 taps x 2 kgrp x 4 N-frags
constexpr int HF_BODY_SZ     = 16 * HF_BODY_LAYER;   // 16 layers
constexpr int HF_TAIL        = HF_BODY + HF_BODY_SZ; // 598016
constexpr int HF_TAIL_EXPERT = 18 * 512;             // 9 taps x 2 kgrp x 1 N-frag
constexpr int HF_TOTAL       = HF_TAIL + 4 * HF_TAIL_EXPERT; // 634880

// dynamic LDS carve (halves): act0 65536 | act1 65536 | wst 18432 | simg 4096 | zer 64
constexpr int SMEM_BYTES = (65536 + 65536 + 18432 + 4096 + 64) * 2; // 307328
// byte offset of wst inside dynamic LDS (sr_main has no static LDS -> dyn base = 0)
constexpr unsigned WST_LDS = 65536u * 2u * 2u;       // 262144

// =====================================================================
// CDNA5 async global->LDS copy (ASYNCcnt-tracked DMA, no VGPR payload)
// =====================================================================
__device__ __forceinline__ void async_b128(unsigned lds_off, const void* gptr) {
  asm volatile("global_load_async_to_lds_b128 %0, %1, off"
               :: "v"(lds_off), "v"((unsigned long long)gptr) : "memory");
}
__device__ __forceinline__ void wait_async0() {
  asm volatile("s_wait_asynccnt 0x0" ::: "memory");
}

// =====================================================================
// zero the output (invalid / capacity-dropped images must be 0)
// =====================================================================
__global__ void zero_out(float4* __restrict__ out4) {
  size_t i = (size_t)blockIdx.x * 256 + threadIdx.x;
  float4 z; z.x = 0.f; z.y = 0.f; z.z = 0.f; z.w = 0.f;
  out4[i] = z;
}

// =====================================================================
// classifier: conv3->16 + relu + spatial mean + FC(16->4) + argmax
// =====================================================================
__global__ __launch_bounds__(256) void cls_kernel(
    const float* __restrict__ in, const float* __restrict__ w1,
    const float* __restrict__ b1, const float* __restrict__ wfc,
    const float* __restrict__ bfc, int* __restrict__ expert) {
  __shared__ float sImg[3 * 1024];
  __shared__ float sW[27 * 16];   // [ci*9+tap][co]
  __shared__ float sRed[16];
  __shared__ float sB1[16];
  int n = blockIdx.x, tid = threadIdx.x;
  for (int i = tid; i < 3072; i += 256) sImg[i] = in[(size_t)n * 3072 + i];
  for (int i = tid; i < 432; i += 256) {
    int co = i & 15, k = i >> 4;          // k = ci*9+tap
    sW[i] = w1[(co * 3 + (k / 9)) * 9 + (k % 9)];
  }
  if (tid < 16) { sRed[tid] = 0.f; sB1[tid] = b1[tid]; }
  __syncthreads();

  float acc[16];
#pragma unroll
  for (int c = 0; c < 16; ++c) acc[c] = 0.f;
  for (int q = 0; q < 4; ++q) {
    int p = tid + q * 256, y = p >> 5, x = p & 31;
    float pacc[16];
#pragma unroll
    for (int c = 0; c < 16; ++c) pacc[c] = sB1[c];
    for (int ci = 0; ci < 3; ++ci) {
#pragma unroll
      for (int t = 0; t < 9; ++t) {
        int yy = y + t / 3 - 1, xx = x + t % 3 - 1;
        float v = ((unsigned)yy < 32u && (unsigned)xx < 32u)
                      ? sImg[ci * 1024 + (yy << 5) + xx] : 0.f;
        const float* wr = &sW[(ci * 9 + t) * 16];
#pragma unroll
        for (int c = 0; c < 16; ++c) pacc[c] += v * wr[c];
      }
    }
#pragma unroll
    for (int c = 0; c < 16; ++c) acc[c] += fmaxf(pacc[c], 0.f);
  }
#pragma unroll
  for (int c = 0; c < 16; ++c) atomicAdd(&sRed[c], acc[c]);
  __syncthreads();
  if (tid == 0) {
    float best = -3.4e38f; int be = 0;
    for (int e = 0; e < 4; ++e) {
      float s = bfc[e];
      for (int c = 0; c < 16; ++c) s += (sRed[c] * (1.f / 1024.f)) * wfc[e * 16 + c];
      if (s > best) { best = s; be = e; }
    }
    expert[n] = be;
  }
}

// =====================================================================
// routing: ordered cumsum with capacity (matches reference semantics)
// =====================================================================
__global__ void route_kernel(const int* __restrict__ expert,
                             int* __restrict__ wlist, int* __restrict__ cnt) {
  if (threadIdx.x == 0 && blockIdx.x == 0) {
    int c[4] = {0, 0, 0, 0}; int total = 0;
    for (int n = 0; n < NIMG; ++n) {
      int e = expert[n];
      if (c[e] < CAP) wlist[total++] = (n << 3) | e;
      c[e]++;
    }
    *cnt = total;
  }
}

// =====================================================================
// pre-pack all conv weights (f32) into f16 WMMA B-fragment layout.
// B 32x16 (KxN) frag: lane = n + 16*hi, half j -> K = (j<8 ? j : j+8) + 8*hi; 1KB per frag.
// =====================================================================
__global__ __launch_bounds__(256) void prep_weights(
    const float* __restrict__ head_w, const float* __restrict__ body_w,
    const float* __restrict__ tail_w, _Float16* __restrict__ wout) {
  int idx = blockIdx.x * 256 + threadIdx.x;
  if (idx >= HF_TOTAL) return;
  float val = 0.f;
  if (idx < HF_BODY) {               // head: K = tap*3 + ci (27 used, pad 32)
    int e = idx / 2048, r = idx % 2048;
    int nt = r / 512, w = r % 512;
    int lane = w / 16, j = w % 16;
    int K = (j < 8 ? j : j + 8) + ((lane >= 16) ? 8 : 0);
    int cout = nt * 16 + (lane & 15);
    if (K < 27) {
      int tap = K / 3, ci = K % 3;
      val = head_w[((e * 64 + cout) * 3 + ci) * 9 + tap];
    }
  } else if (idx < HF_TAIL) {        // body: fragIdx = tap*8 + kg*4 + nt
    int t = idx - HF_BODY;
    int layer = t / HF_BODY_LAYER;   // e*4+b
    int r = t % HF_BODY_LAYER;
    int fi = r / 512, w = r % 512;
    int tap = fi >> 3, kg = (fi >> 2) & 1, nt = fi & 3;
    int lane = w / 16, j = w % 16;
    int K = (j < 8 ? j : j + 8) + ((lane >= 16) ? 8 : 0);
    int cin = kg * 32 + K;
    int cout = nt * 16 + (lane & 15);
    val = body_w[((layer * 64 + cout) * 64 + cin) * 9 + tap];
  } else {                           // tail: fragIdx = tap*2 + kg, 12 couts padded to 16
    int t = idx - HF_TAIL;
    int e = t / HF_TAIL_EXPERT;
    int r = t % HF_TAIL_EXPERT;
    int fi = r / 512, w = r % 512;
    int tap = fi >> 1, kg = fi & 1;
    int lane = w / 16, j = w % 16;
    int K = (j < 8 ? j : j + 8) + ((lane >= 16) ? 8 : 0);
    int cin = kg * 32 + K;
    int cout = lane & 15;
    if (cout < 12) val = tail_w[((e * 12 + cout) * 64 + cin) * 9 + tap];
  }
  wout[idx] = (_Float16)val;
}

// =====================================================================
// main SR kernel helpers
// =====================================================================
// A 16x32 f16 frag: lane lo/hi hold same pixel row M; per lane halves
// j=0..7 -> K = j + 8*hi, j=8..15 -> K = j+8 + 8*hi. Two 16B LDS loads.
__device__ __forceinline__ v16h load_a(const _Float16* act, const _Float16* zer,
                                       int mtBase, int tap, int hbase, int l15) {
  int p = mtBase + l15;
  int y = p >> 5, x = p & 31;
  int yy = y + tap / 3 - 1;
  int xx = x + tap % 3 - 1;
  bool ok = ((unsigned)yy < 32u) && ((unsigned)xx < 32u);
  const _Float16* s = ok ? (act + (((yy << 5) + xx) << 6) + hbase) : zer;
  union { v16h v; v8h h[2]; } u;
  u.h[0] = *(const v8h*)s;         // K j=0..7  (cin hbase..+7)
  u.h[1] = *(const v8h*)(s + 16);  // K j=8..15 (cin hbase+16..+23)
  return u.v;
}

// head conv as one padded K=32 WMMA step (K = tap*3+ci, 27 used)
__device__ __forceinline__ void head_mm(const _Float16* simg, const _Float16* zer,
                                        const _Float16* __restrict__ whead,
                                        int mtBase, int lane, v8f* acc) {
  int laneHi = lane >> 4, l15 = lane & 15;
  int p = mtBase + l15, y = p >> 5, x = p & 31;
  union { v16h v; _Float16 s[16]; } ua;
#pragma unroll
  for (int j = 0; j < 16; ++j) {
    int K = (j < 8 ? j : j + 8) + (laneHi << 3);
    int tap = K / 3, ci = K - tap * 3;
    int yy = y + tap / 3 - 1, xx = x + (tap - (tap / 3) * 3) - 1;
    bool ok = (K < 27) && ((unsigned)yy < 32u) && ((unsigned)xx < 32u);
    const _Float16* sp = ok ? (simg + (((yy << 5) + xx) << 2) + ci) : zer;
    ua.s[j] = *sp;
  }
#pragma unroll
  for (int nt = 0; nt < 4; ++nt) {
    v16h b = *(const v16h*)(whead + nt * 512 + lane * 16);
    acc[nt] = __builtin_amdgcn_wmma_f32_16x16x32_f16(
        false, ua.v, false, b, (short)0, acc[nt], false, false);
  }
}

// one 64->64 conv layer with bias+relu: src(LDS) -> dst(LDS).
// Two cout-half passes; weights staged via async global->LDS DMA.
__device__ __forceinline__ void conv64(const _Float16* __restrict__ src,
                                       _Float16* __restrict__ dst,
                                       _Float16* wst, const _Float16* zer,
                                       const _Float16* __restrict__ wlayer,
                                       const float* __restrict__ bias, int tid) {
  int lane = tid & 31, wid = tid >> 5;
  int laneHi = lane >> 4, l15 = lane & 15;
  for (int nh = 0; nh < 2; ++nh) {
    __syncthreads();
    // stage 18 chunks of 1024 halves (2 N-frags each) -> contiguous in wst
    for (int i = tid; i < 2304; i += 256) {
      int gh = i * 8;
      int tk = gh >> 10, off = gh & 1023;
      async_b128(WST_LDS + (unsigned)gh * 2u,
                 (const void*)(wlayer + (tk * 4 + nh * 2) * 512 + off));
    }
    wait_async0();
    __syncthreads();
    for (int mi = 0; mi < 8; ++mi) {
      int mtBase = (wid * 8 + mi) * 16;
      v8f acc0 = {}; v8f acc1 = {};
#pragma unroll
      for (int tap = 0; tap < 9; ++tap) {
#pragma unroll
        for (int kg = 0; kg < 2; ++kg) {
          v16h a = load_a(src, zer, mtBase, tap, kg * 32 + (laneHi << 3), l15);
          const _Float16* wp = wst + (tap * 2 + kg) * 1024 + lane * 16;
          v16h b0 = *(const v16h*)wp;
          v16h b1 = *(const v16h*)(wp + 512);
          acc0 = __builtin_amdgcn_wmma_f32_16x16x32_f16(
              false, a, false, b0, (short)0, acc0, false, false);
          acc1 = __builtin_amdgcn_wmma_f32_16x16x32_f16(
              false, a, false, b1, (short)0, acc1, false, false);
        }
      }
      int co0 = nh * 32 + l15;
      float bv0 = bias[co0], bv1 = bias[co0 + 16];
#pragma unroll
      for (int v = 0; v < 8; ++v) {
        int p = mtBase + v + (laneHi << 3);
        float r0 = fmaxf(acc0[v] + bv0, 0.f);
        float r1 = fmaxf(acc1[v] + bv1, 0.f);
        dst[(p << 6) + co0] = (_Float16)r0;
        dst[(p << 6) + co0 + 16] = (_Float16)r1;
      }
    }
  }
}

// =====================================================================
// main kernel: one workgroup = one routed image, whole net in LDS
// =====================================================================
__global__ __launch_bounds__(256, 1) void sr_main(
    const float* __restrict__ in, const float* __restrict__ head_b,
    const float* __restrict__ body_b, const float* __restrict__ tail_b,
    float* __restrict__ out, const int* __restrict__ wscnt,
    const int* __restrict__ wlist, const _Float16* __restrict__ whalf) {
  int w = blockIdx.x;
  if (w >= wscnt[0]) return;
  int packed = wlist[w];
  int img = packed >> 3, e = packed & 7;

  extern __shared__ char dsm[];
  _Float16* act0 = (_Float16*)dsm;          // [px][64]
  _Float16* act1 = act0 + 65536;
  _Float16* wst  = act1 + 65536;            // 18432 halves @ LDS byte 262144
  _Float16* simg = wst + 18432;             // [px][4] (ch3 = 0)
  _Float16* zer  = simg + 4096;             // 64-half zero page

  int tid = threadIdx.x, lane = tid & 31, wid = tid >> 5;
  int laneHi = lane >> 4, l15 = lane & 15;

  // load input image f32 -> f16 LDS
  for (int i = tid; i < 3072; i += 256) {
    int c = i >> 10, px = i & 1023;
    simg[(px << 2) + c] = (_Float16)in[(size_t)img * 3072 + i];
  }
  for (int i = tid; i < 1024; i += 256) simg[(i << 2) + 3] = (_Float16)0.f;
  if (tid < 64) zer[tid] = (_Float16)0.f;

  // warm L2 with this expert's body+tail fragments (288KB + 18KB)
  {
    const char* pf = (const char*)(whalf + HF_BODY + (size_t)(e * 4) * HF_BODY_LAYER);
    for (int i = tid; i < 1152; i += 256)
      __builtin_prefetch(pf + (size_t)i * 256, 0, 1);
    const char* pt = (const char*)(whalf + HF_TAIL + e * HF_TAIL_EXPERT);
    if (tid < 72) __builtin_prefetch(pt + (size_t)tid * 256, 0, 1);
  }
  __syncthreads();

  // ---- HEAD -> act0 (h0, no relu) ----
  const _Float16* wh = whalf + HF_HEAD + e * 2048;
  for (int mi = 0; mi < 8; ++mi) {
    int mtBase = (wid * 8 + mi) * 16;
    v8f acc[4] = {};
    head_mm(simg, zer, wh, mtBase, lane, acc);
#pragma unroll
    for (int nt = 0; nt < 4; ++nt) {
      int co = nt * 16 + l15;
      float bv = head_b[e * 64 + co];
#pragma unroll
      for (int v = 0; v < 8; ++v) {
        int p = mtBase + v + (laneHi << 3);
        act0[(p << 6) + co] = (_Float16)(acc[nt][v] + bv);
      }
    }
  }
  __syncthreads();

  // ---- BODY x4 (ping-pong; h0 recomputed later, so act0 is reusable) ----
  const _Float16* wb = whalf + HF_BODY;
  conv64(act0, act1, wst, zer, wb + (size_t)(e * 4 + 0) * HF_BODY_LAYER,
         body_b + (e * 4 + 0) * 64, tid);
  __syncthreads();
  conv64(act1, act0, wst, zer, wb + (size_t)(e * 4 + 1) * HF_BODY_LAYER,
         body_b + (e * 4 + 1) * 64, tid);
  __syncthreads();
  conv64(act0, act1, wst, zer, wb + (size_t)(e * 4 + 2) * HF_BODY_LAYER,
         body_b + (e * 4 + 2) * 64, tid);
  __syncthreads();
  conv64(act1, act0, wst, zer, wb + (size_t)(e * 4 + 3) * HF_BODY_LAYER,
         body_b + (e * 4 + 3) * 64, tid);
  __syncthreads();

  // ---- SKIP: act0 += h0 (recompute cheap head conv, in place) ----
  for (int mi = 0; mi < 8; ++mi) {
    int mtBase = (wid * 8 + mi) * 16;
    v8f acc[4] = {};
    head_mm(simg, zer, wh, mtBase, lane, acc);
#pragma unroll
    for (int nt = 0; nt < 4; ++nt) {
      int co = nt * 16 + l15;
      float bv = head_b[e * 64 + co];
#pragma unroll
      for (int v = 0; v < 8; ++v) {
        int p = mtBase + v + (laneHi << 3);
        float cur = (float)act0[(p << 6) + co];
        act0[(p << 6) + co] = (_Float16)(cur + acc[nt][v] + bv);
      }
    }
  }
  __syncthreads();

  // ---- TAIL: 64->12 conv + bias + pixel-shuffle scatter to global ----
  const _Float16* wt = whalf + HF_TAIL + e * HF_TAIL_EXPERT;
  for (int i = tid; i < 2304; i += 256)
    async_b128(WST_LDS + (unsigned)i * 16u, (const void*)(wt + i * 8));
  wait_async0();
  __syncthreads();
  for (int mi = 0; mi < 8; ++mi) {
    int mtBase = (wid * 8 + mi) * 16;
    v8f acc = {};
#pragma unroll
    for (int tap = 0; tap < 9; ++tap) {
#pragma unroll
      for (int kg = 0; kg < 2; ++kg) {
        v16h a = load_a(act0, zer, mtBase, tap, kg * 32 + (laneHi << 3), l15);
        v16h b = *(const v16h*)(wst + (tap * 2 + kg) * 512 + lane * 16);
        acc = __builtin_amdgcn_wmma_f32_16x16x32_f16(
            false, a, false, b, (short)0, acc, false, false);
      }
    }
    int t = l15;
    if (t < 12) {
      float bv = tail_b[e * 12 + t];
      int c = t >> 2, sy = (t >> 1) & 1, sx = t & 1;
#pragma unroll
      for (int v = 0; v < 8; ++v) {
        int p = mtBase + v + (laneHi << 3);
        int y = p >> 5, x = p & 31;
        out[(((size_t)img * 3 + c) << 12) + (((y << 1) + sy) << 6) + ((x << 1) + sx)]
            = acc[v] + bv;
      }
    }
  }
}

// =====================================================================
extern "C" void kernel_launch(void* const* d_in, const int* in_sizes, int n_in,
                              void* d_out, int out_size, void* d_ws, size_t ws_size,
                              hipStream_t stream) {
  const float* in      = (const float*)d_in[0];
  const float* cls_w1  = (const float*)d_in[1];
  const float* cls_b1  = (const float*)d_in[2];
  const float* cls_wfc = (const float*)d_in[3];
  const float* cls_bfc = (const float*)d_in[4];
  const float* head_w  = (const float*)d_in[5];
  const float* head_b  = (const float*)d_in[6];
  const float* body_w  = (const float*)d_in[7];
  const float* body_b  = (const float*)d_in[8];
  const float* tail_w  = (const float*)d_in[9];
  const float* tail_b  = (const float*)d_in[10];
  float* out = (float*)d_out;

  char* ws = (char*)d_ws;
  int* cnt        = (int*)(ws + WS_COUNT);
  int* wlist      = (int*)(ws + WS_WORKLIST);
  int* expert     = (int*)(ws + WS_EXPERT);
  _Float16* whalf = (_Float16*)(ws + WS_WHALF);

  (void)hipFuncSetAttribute((const void*)sr_main,
                            hipFuncAttributeMaxDynamicSharedMemorySize, SMEM_BYTES);

  zero_out<<<24576, 256, 0, stream>>>((float4*)out);                 // 2048*3*64*64 floats
  cls_kernel<<<NIMG, 256, 0, stream>>>(in, cls_w1, cls_b1, cls_wfc, cls_bfc, expert);
  route_kernel<<<1, 32, 0, stream>>>(expert, wlist, cnt);
  prep_weights<<<HF_TOTAL / 256, 256, 0, stream>>>(head_w, body_w, tail_w, whalf);
  sr_main<<<NIMG, 256, SMEM_BYTES, stream>>>(in, head_b, body_b, tail_b, out,
                                             cnt, wlist, whalf);
}
